// MultiHeadAttention_46608985096332
// MI455X (gfx1250) — matrix operand order repaired and checked by reading, not compiled, not run on hardware
//
#include <hip/hip_runtime.h>
#include <hip/hip_bf16.h>
#include <math.h>
#include <stdint.h>

// Problem constants (match reference)
#define B_    4
#define S_    2048
#define HID_  1280
#define NH_   16
#define NKV_  4
#define HD_   80
#define DPAD_ 96          // HD padded to 3*32 for bf16 WMMA K-steps
#define QKVN_ 1920        // (NH + 2*NKV) * HD
#define NEGINF (-3.0e38f)

typedef __bf16 bf16_t;
typedef __attribute__((ext_vector_type(16))) __bf16 v16bf;
typedef __attribute__((ext_vector_type(8)))  __bf16 v8bf;
typedef __attribute__((ext_vector_type(8)))  float  v8f;

// ---------------------------------------------------------------------------
// WMMA fragment helpers (bf16 16x16x32, wave32).
// A (16xK=32, 16-bit): lanes 0-15 hold row M=lane, K pairs 0..7 in v0-3 and
// 16..23 in v4-7; lanes 16-31 hold same rows, K 8..15 / 24..31.
// B (K=32x16, 16-bit): lane holds column N=lane&15; same K order as A:
// lanes 0-15 K 0..7 / 16..23, lanes 16-31 K 8..15 / 24..31.
// ---------------------------------------------------------------------------
__device__ __forceinline__ v16bf load_a_frag(const bf16_t* __restrict__ rowptr,
                                             int k0, int hi) {
  v8bf a0 = *(const v8bf*)(rowptr + k0 + 8 * hi);
  v8bf a1 = *(const v8bf*)(rowptr + k0 + 16 + 8 * hi);
  v16bf r;
#pragma unroll
  for (int i = 0; i < 8; ++i) { r[i] = a0[i]; r[i + 8] = a1[i]; }
  return r;
}

__device__ __forceinline__ v16bf load_b_frag(const bf16_t* __restrict__ rowptr,
                                             int k0, int hi) {
  v8bf b0 = *(const v8bf*)(rowptr + k0 + 8 * hi);
  v8bf b1 = *(const v8bf*)(rowptr + k0 + 16 + 8 * hi);
  v16bf r;
#pragma unroll
  for (int i = 0; i < 8; ++i) { r[i] = b0[i]; r[i + 8] = b1[i]; }
  return r;
}

__device__ __forceinline__ v8f wmma_bf16(v16bf a, v16bf b, v8f c) {
  return __builtin_amdgcn_wmma_f32_16x16x32_bf16(false, a, false, b,
                                                 (short)0, c, false, false);
}
// Variant with reuse-A hint: legal when the next instruction is an identical
// WMMA consuming the same A operand (our 4-wide N tiling below).
__device__ __forceinline__ v8f wmma_bf16_ra(v16bf a, v16bf b, v8f c) {
  return __builtin_amdgcn_wmma_f32_16x16x32_bf16(false, a, false, b,
                                                 (short)0, c, true, false);
}

// ---------------------------------------------------------------------------
// Prep kernels
// ---------------------------------------------------------------------------
__global__ void k_scale(const float* __restrict__ scaling,
                        float* __restrict__ scale) {
  int d = threadIdx.x;
  if (d < HD_) {
    float x = scaling[d];
    float sp = (x > 20.f) ? x : log1pf(__expf(x));  // softplus
    scale[d] = 1.442695041f * rsqrtf((float)HD_) * sp;
  }
}

__global__ void k_cvt_bf16(const float* __restrict__ src,
                           bf16_t* __restrict__ dst, int n) {
  for (int i = blockIdx.x * blockDim.x + threadIdx.x; i < n;
       i += gridDim.x * blockDim.x)
    dst[i] = (bf16_t)src[i];
}

__global__ void k_zero_u32(uint32_t* __restrict__ p, int n) {
  for (int i = blockIdx.x * blockDim.x + threadIdx.x; i < n;
       i += gridDim.x * blockDim.x)
    p[i] = 0u;
}

// ---------------------------------------------------------------------------
// QKV projection: [B*S,1280] x [1920,1280]^T.
// Each wave computes a 16x64 C tile (4 N-subtiles): A fragment loaded once
// per k-step and fed to 4 WMMAs with reuse-A hints.
// Epilogue scatters into padded Q (scaled), padded K, transposed V.
// ---------------------------------------------------------------------------
__global__ void __launch_bounds__(256) k_qkv_gemm(
    const bf16_t* __restrict__ hbf, const bf16_t* __restrict__ wbf,
    const float* __restrict__ qkv_b, const float* __restrict__ scale,
    bf16_t* __restrict__ Qb, bf16_t* __restrict__ Kb,
    bf16_t* __restrict__ Vt) {
  const int lane = threadIdx.x & 31;
  const int hi = lane >> 4;
  const int lm = lane & 15;
  const int wid = (blockIdx.x * blockDim.x + threadIdx.x) >> 5;
  const int tilesNg = QKVN_ / 64;  // 30 groups of 4 N-tiles
  const int tg = wid % tilesNg;
  const int tm = wid / tilesNg;
  const int m0 = tm * 16, n0 = tg * 64;

  const bf16_t* arow = hbf + (size_t)(m0 + lm) * HID_;
  const bf16_t* brow0 = wbf + (size_t)(n0 + lm) * HID_;
  const bf16_t* brow1 = wbf + (size_t)(n0 + 16 + lm) * HID_;
  const bf16_t* brow2 = wbf + (size_t)(n0 + 32 + lm) * HID_;
  const bf16_t* brow3 = wbf + (size_t)(n0 + 48 + lm) * HID_;

  v8f c0 = {0.f, 0.f, 0.f, 0.f, 0.f, 0.f, 0.f, 0.f};
  v8f c1 = c0, c2 = c0, c3 = c0;
  for (int k0 = 0; k0 < HID_; k0 += 32) {
    // WGP-scope prefetch (high locality -> near caches), ~8 k-steps ahead
    __builtin_prefetch(arow + k0 + 256, 0, 3);
    __builtin_prefetch(brow0 + k0 + 256, 0, 3);
    v16bf a = load_a_frag(arow, k0, hi);
    c0 = wmma_bf16_ra(a, load_b_frag(brow0, k0, hi), c0);
    c1 = wmma_bf16_ra(a, load_b_frag(brow1, k0, hi), c1);
    c2 = wmma_bf16_ra(a, load_b_frag(brow2, k0, hi), c2);
    c3 = wmma_bf16   (a, load_b_frag(brow3, k0, hi), c3);
  }

  v8f cs[4] = {c0, c1, c2, c3};
#pragma unroll
  for (int j = 0; j < 4; ++j) {
    const int n = n0 + j * 16 + lm;
    const float bias = qkv_b[n];
#pragma unroll
    for (int r = 0; r < 8; ++r) {
      int m = m0 + r + 8 * hi;
      int bi = m / S_, s = m % S_;
      float v = cs[j][r] + bias;
      if (n < NH_ * HD_) {                       // Q: scale, pad layout
        int hh = n / HD_, d = n % HD_;
        v *= scale[d];
        Qb[(((size_t)bi * NH_ + hh) * S_ + s) * DPAD_ + d] = (bf16_t)v;
      } else if (n < NH_ * HD_ + NKV_ * HD_) {   // K: pad layout
        int nn = n - NH_ * HD_;
        int hh = nn / HD_, d = nn % HD_;
        Kb[(((size_t)bi * NKV_ + hh) * S_ + s) * DPAD_ + d] = (bf16_t)v;
      } else {                                   // V: transposed [d][s]
        int nn = n - NH_ * HD_ - NKV_ * HD_;
        int hh = nn / HD_, d = nn % HD_;
        Vt[(((size_t)bi * NKV_ + hh) * HD_ + d) * S_ + s] = (bf16_t)v;
      }
    }
  }
}

// ---------------------------------------------------------------------------
// Flash attention: one 16-row q tile per wave, 32-key chunks, causal bound.
// ---------------------------------------------------------------------------
__global__ void __launch_bounds__(256) k_attn(
    const bf16_t* __restrict__ Qb, const bf16_t* __restrict__ Kb,
    const bf16_t* __restrict__ Vt, bf16_t* __restrict__ Ab) {
  __shared__ __attribute__((aligned(32))) bf16_t Plds[8][16][48];

  const int lane = threadIdx.x & 31;
  const int hi = lane >> 4;
  const int lm = lane & 15;
  const int w = threadIdx.x >> 5;
  const int wid = (blockIdx.x * blockDim.x + threadIdx.x) >> 5;
  const int qtiles = S_ / 16;  // 128
  const int qt = wid % qtiles;
  const int bh = wid / qtiles;
  const int h = bh % NH_;
  const int bi = bh / NH_;
  const int kvh = h / (NH_ / NKV_);
  const int q0 = qt * 16;

  // Q fragments for this tile (3 K-chunks of 32 over padded D=96)
  const bf16_t* qrow = Qb + (((size_t)bi * NH_ + h) * S_ + q0 + lm) * DPAD_;
  v16bf qa0 = load_a_frag(qrow, 0, hi);
  v16bf qa1 = load_a_frag(qrow, 32, hi);
  v16bf qa2 = load_a_frag(qrow, 64, hi);

  // Per-lane row stats for the 8 rows this half-group owns (r + 8*hi)
  float mrow[8], lrow[8];
#pragma unroll
  for (int r = 0; r < 8; ++r) { mrow[r] = NEGINF; lrow[r] = 0.f; }
  v8f acc[5];
#pragma unroll
  for (int t = 0; t < 5; ++t) acc[t] = (v8f){0.f, 0.f, 0.f, 0.f, 0.f, 0.f, 0.f, 0.f};

  const bf16_t* Kbase = Kb + ((size_t)bi * NKV_ + kvh) * (size_t)S_ * DPAD_;
  const bf16_t* Vbase = Vt + ((size_t)bi * NKV_ + kvh) * (size_t)HD_ * S_;

  const int kend = q0 + 16;  // causal: keys < q0+16 needed
  for (int kb = 0; kb < kend; kb += 32) {
    // prefetch next chunk's K rows into near caches (speculative; OOB dropped)
    __builtin_prefetch(Kbase + (size_t)(kb + 32 + lm) * DPAD_, 0, 3);

    // ---- scores for keys [kb, kb+32) : two 16-wide C tiles -------------
    v8f s0 = {0.f, 0.f, 0.f, 0.f, 0.f, 0.f, 0.f, 0.f};
    v8f s1 = {0.f, 0.f, 0.f, 0.f, 0.f, 0.f, 0.f, 0.f};
    {
      const bf16_t* krow = Kbase + (size_t)(kb + lm) * DPAD_;
      s0 = wmma_bf16(qa0, load_b_frag(krow, 0, hi), s0);
      s0 = wmma_bf16(qa1, load_b_frag(krow, 32, hi), s0);
      s0 = wmma_bf16(qa2, load_b_frag(krow, 64, hi), s0);
    }
    {
      const bf16_t* krow = Kbase + (size_t)(kb + 16 + lm) * DPAD_;
      s1 = wmma_bf16(qa0, load_b_frag(krow, 0, hi), s1);
      s1 = wmma_bf16(qa1, load_b_frag(krow, 32, hi), s1);
      s1 = wmma_bf16(qa2, load_b_frag(krow, 64, hi), s1);
    }

    // ---- causal mask + online softmax ----------------------------------
    const int k0g = kb + lm;
    const int k1g = kb + 16 + lm;
    float f[8];
#pragma unroll
    for (int r = 0; r < 8; ++r) {
      const int qg = q0 + r + 8 * hi;
      float a0 = (k0g <= qg) ? s0[r] : NEGINF;
      float a1 = (k1g <= qg) ? s1[r] : NEGINF;
      float mx = fmaxf(a0, a1);
      mx = fmaxf(mx, __shfl_xor(mx, 1));
      mx = fmaxf(mx, __shfl_xor(mx, 2));
      mx = fmaxf(mx, __shfl_xor(mx, 4));
      mx = fmaxf(mx, __shfl_xor(mx, 8));
      float mnew = fmaxf(mrow[r], mx);
      float p0 = __expf(a0 - mnew);
      float p1 = __expf(a1 - mnew);
      float rs = p0 + p1;
      rs += __shfl_xor(rs, 1);
      rs += __shfl_xor(rs, 2);
      rs += __shfl_xor(rs, 4);
      rs += __shfl_xor(rs, 8);
      f[r] = __expf(mrow[r] - mnew);
      lrow[r] = lrow[r] * f[r] + rs;
      mrow[r] = mnew;
      // stash probabilities (C-layout -> LDS, row-major 16x32, stride 48)
      Plds[w][r + 8 * hi][lm] = (bf16_t)p0;
      Plds[w][r + 8 * hi][16 + lm] = (bf16_t)p1;
      // rescale running accumulators
#pragma unroll
      for (int t = 0; t < 5; ++t) acc[t][r] *= f[r];
    }

    // ---- P tile (16x32) as A-fragment from LDS -------------------------
    const bf16_t* prow = &Plds[w][lm][0];
    v8bf x0 = *(const v8bf*)(prow + 8 * hi);
    v8bf x1 = *(const v8bf*)(prow + 16 + 8 * hi);
    v16bf pa;
#pragma unroll
    for (int i = 0; i < 8; ++i) { pa[i] = x0[i]; pa[i + 8] = x1[i]; }

    // ---- PV: 5 output d-tiles, contraction over 32 keys ----------------
#pragma unroll
    for (int t = 0; t < 5; ++t) {
      const bf16_t* vrow = Vbase + (size_t)(t * 16 + lm) * S_;
      acc[t] = (t < 4) ? wmma_bf16_ra(pa, load_b_frag(vrow, kb, hi), acc[t])
                       : wmma_bf16(pa, load_b_frag(vrow, kb, hi), acc[t]);
    }
  }

  // ---- finalize: divide by l, store bf16 attention output --------------
#pragma unroll
  for (int r = 0; r < 8; ++r) {
    float inv = 1.f / lrow[r];
    int s = q0 + r + 8 * hi;
#pragma unroll
    for (int t = 0; t < 5; ++t) {
      float o = acc[t][r] * inv;
      Ab[((size_t)bi * S_ + s) * HID_ + h * HD_ + t * 16 + lm] = (bf16_t)o;
    }
  }
}

// ---------------------------------------------------------------------------
// Output projection: [B*S,1280] x [1280,1280]^T + bias -> fp32 out.
// 16x64 C tile per wave with reuse-A, as in k_qkv_gemm.
// ---------------------------------------------------------------------------
__global__ void __launch_bounds__(256) k_oproj(
    const bf16_t* __restrict__ Ab, const bf16_t* __restrict__ wo,
    const float* __restrict__ o_b, float* __restrict__ out) {
  const int lane = threadIdx.x & 31;
  const int hi = lane >> 4;
  const int lm = lane & 15;
  const int wid = (blockIdx.x * blockDim.x + threadIdx.x) >> 5;
  const int tilesNg = HID_ / 64;  // 20 groups of 4 N-tiles
  const int tg = wid % tilesNg;
  const int tm = wid / tilesNg;
  const int m0 = tm * 16, n0 = tg * 64;

  const bf16_t* arow = Ab + (size_t)(m0 + lm) * HID_;
  const bf16_t* brow0 = wo + (size_t)(n0 + lm) * HID_;
  const bf16_t* brow1 = wo + (size_t)(n0 + 16 + lm) * HID_;
  const bf16_t* brow2 = wo + (size_t)(n0 + 32 + lm) * HID_;
  const bf16_t* brow3 = wo + (size_t)(n0 + 48 + lm) * HID_;

  v8f c0 = {0.f, 0.f, 0.f, 0.f, 0.f, 0.f, 0.f, 0.f};
  v8f c1 = c0, c2 = c0, c3 = c0;
  for (int k0 = 0; k0 < HID_; k0 += 32) {
    __builtin_prefetch(arow + k0 + 256, 0, 3);
    __builtin_prefetch(brow0 + k0 + 256, 0, 3);
    v16bf a = load_a_frag(arow, k0, hi);
    c0 = wmma_bf16_ra(a, load_b_frag(brow0, k0, hi), c0);
    c1 = wmma_bf16_ra(a, load_b_frag(brow1, k0, hi), c1);
    c2 = wmma_bf16_ra(a, load_b_frag(brow2, k0, hi), c2);
    c3 = wmma_bf16   (a, load_b_frag(brow3, k0, hi), c3);
  }

  v8f cs[4] = {c0, c1, c2, c3};
#pragma unroll
  for (int j = 0; j < 4; ++j) {
    const int n = n0 + j * 16 + lm;
    const float bias = o_b[n];
#pragma unroll
    for (int r = 0; r < 8; ++r) {
      int m = m0 + r + 8 * hi;
      out[(size_t)m * HID_ + n] = cs[j][r] + bias;
    }
  }
}

// ---------------------------------------------------------------------------
// Host launch
// ---------------------------------------------------------------------------
extern "C" void kernel_launch(void* const* d_in, const int* in_sizes, int n_in,
                              void* d_out, int out_size, void* d_ws,
                              size_t ws_size, hipStream_t stream) {
  (void)in_sizes; (void)n_in; (void)out_size; (void)ws_size;
  const float* hidden  = (const float*)d_in[0];
  // d_in[1] = mask (additive causal) — reproduced analytically in-kernel
  const float* scaling = (const float*)d_in[2];
  const float* qkv_w   = (const float*)d_in[3];
  const float* qkv_b   = (const float*)d_in[4];
  const float* o_w     = (const float*)d_in[5];
  const float* o_b     = (const float*)d_in[6];
  float* out = (float*)d_out;

  // Workspace carve-up (256B aligned)
  char* ws = (char*)d_ws;
  size_t off = 0;
  auto carve = [&](size_t bytes) {
    size_t o = off;
    off += (bytes + 255) & ~(size_t)255;
    return (void*)(ws + o);
  };
  const int n_h   = B_ * S_ * HID_;        // 10,485,760
  const int n_wq  = QKVN_ * HID_;          //  2,457,600
  const int n_wo  = HID_ * HID_;           //  1,638,400
  const int n_Qb  = B_ * NH_ * S_ * DPAD_; // 12,582,912
  const int n_Kb  = B_ * NKV_ * S_ * DPAD_;//  3,145,728
  const int n_Vt  = B_ * NKV_ * HD_ * S_;  //  2,621,440
  const int n_Ab  = B_ * S_ * HID_;        // 10,485,760

  float*  scale = (float*) carve(HD_ * sizeof(float));
  bf16_t* hbf   = (bf16_t*)carve((size_t)n_h  * 2);
  bf16_t* wqkv  = (bf16_t*)carve((size_t)n_wq * 2);
  bf16_t* wo    = (bf16_t*)carve((size_t)n_wo * 2);
  bf16_t* Qb    = (bf16_t*)carve((size_t)n_Qb * 2);  // Qb/Kb adjacent
  bf16_t* Kb    = (bf16_t*)carve((size_t)n_Kb * 2);
  bf16_t* Vt    = (bf16_t*)carve((size_t)n_Vt * 2);
  bf16_t* Ab    = (bf16_t*)carve((size_t)n_Ab * 2);

  // 1) softplus query scale
  k_scale<<<1, 128, 0, stream>>>(scaling, scale);
  // 2) fp32 -> bf16 conversions
  k_cvt_bf16<<<2048, 256, 0, stream>>>(hidden, hbf, n_h);
  k_cvt_bf16<<<2048, 256, 0, stream>>>(qkv_w, wqkv, n_wq);
  k_cvt_bf16<<<2048, 256, 0, stream>>>(o_w, wo, n_wo);
  // 3) zero Q/K padded buffers (contiguous region: Qb then Kb)
  k_zero_u32<<<2048, 256, 0, stream>>>((uint32_t*)Qb, (n_Qb + n_Kb) / 2);
  // 4) QKV projection GEMM (512 M-tiles x 30 N-groups, 8 waves/block)
  k_qkv_gemm<<<(512 * 30) / 8, 256, 0, stream>>>(hbf, wqkv, qkv_b, scale,
                                                 Qb, Kb, Vt);
  // 5) flash attention (B*NH*S/16 = 8192 waves)
  k_attn<<<8192 / 8, 256, 0, stream>>>(Qb, Kb, Vt, Ab);
  // 6) output projection GEMM (512 M-tiles x 20 N-groups)
  k_oproj<<<(512 * 20) / 8, 256, 0, stream>>>(Ab, wo, o_b, out);
}
